// LinearAttention_66907000537730
// MI455X (gfx1250) — matrix operand order repaired and checked by reading, not compiled, not run on hardware
//
#include <hip/hip_runtime.h>
#include <hip/hip_bf16.h>

// ---------------------------------------------------------------------------
// LinearAttention on MI455X (gfx1250, wave32, WMMA).
// GEMMs: v_wmma_f32_16x16x32_bf16, LDS double-buffered via
// global_load_async_to_lds_b128 (ASYNCcnt), ds_load_b128 fragment reads.
// ---------------------------------------------------------------------------

typedef __attribute__((ext_vector_type(16))) __bf16 v16bf;
typedef __attribute__((ext_vector_type(8)))  float  v8f;

#define WMMA_BF16(a, b, c) \
  __builtin_amdgcn_wmma_f32_16x16x32_bf16(false, (a), false, (b), (short)0, (c), false, false)

static constexpr int Bsz  = 4;
static constexpr int Ntok = 4096;
static constexpr int C    = 1024;
static constexpr int H    = 16;
static constexpr int Dh   = 64;
static constexpr int C3   = 3 * C;           // 3072
static constexpr int MTOK = Bsz * Ntok;      // 16384

static constexpr int BM  = 128;              // block tile M
static constexpr int BN  = 128;              // block tile N
static constexpr int BK  = 32;               // K step (one WMMA K-chunk)
static constexpr int LDK = 48;               // padded LDS K stride (96 B, 16B-aligned)

// ---- async copy: global -> LDS (CDNA5, tracked by ASYNCcnt) ---------------
__device__ __forceinline__ void async_b128_to_lds(const __bf16* gp, const __bf16* lp) {
  uint32_t lds  = (uint32_t)(uintptr_t)lp;          // flat LDS addr: low 32 bits = LDS offset
  uint64_t gptr = (uint64_t)(uintptr_t)gp;
  asm volatile("global_load_async_to_lds_b128 %0, %1, off"
               :: "v"(lds), "v"(gptr) : "memory");
}
__device__ __forceinline__ void wait_async0() {
  asm volatile("s_wait_asynccnt 0x0" ::: "memory");
}

// ---- fragment loaders -----------------------------------------------------
// A (16x32 bf16, row-major, leading dim ld). Lanes 0-15: rows 0-15,
// K = {0..7,16..23}; lanes 16-31: same rows, K = {8..15,24..31}.
__device__ __forceinline__ v16bf load_frag_a(const __bf16* base, int ld) {
  const int lane = threadIdx.x & 31;
  const __bf16* p = base + (lane & 15) * ld + ((lane & 16) ? 8 : 0);
  v16bf a;
  ((uint4*)&a)[0] = *(const uint4*)p;
  ((uint4*)&a)[1] = *(const uint4*)(p + 16);
  return a;
}
// B (32x16 bf16). Memory holds B^T: column j contiguous at base+j*ld.
__device__ __forceinline__ v16bf load_frag_b(const __bf16* base, int ld) {
  const int lane = threadIdx.x & 31;
  const __bf16* p = base + (lane & 15) * ld + ((lane & 16) ? 8 : 0);
  v16bf b;
  ((uint4*)&b)[0] = *(const uint4*)p;
  ((uint4*)&b)[1] = *(const uint4*)(p + 16);
  return b;
}

// ---- fp32 -> bf16 conversion ----------------------------------------------
__global__ void __launch_bounds__(256) cvt_f32_bf16(const float* __restrict__ s,
                                                    __bf16* __restrict__ d, int n4) {
  int i = blockIdx.x * blockDim.x + threadIdx.x;
  if (i >= n4) return;
  float4 f = ((const float4*)s)[i];
  union { uint2 u; __bf16 h[4]; } pk;
  pk.h[0] = (__bf16)f.x; pk.h[1] = (__bf16)f.y;
  pk.h[2] = (__bf16)f.z; pk.h[3] = (__bf16)f.w;
  ((uint2*)d)[i] = pk.u;
}

// ---- shared GEMM mainloop body (LDS double-buffered, async staged) --------
// As/Bs: [2][128][LDK]; computes acc[2][4] for this wave (32x64 tile).
// A rows from aG (row-major, ld C), B columns from bG (B^T rows, ld C).
struct GemmAcc { v8f a[2][4]; };

__device__ __forceinline__ void gemm_mainloop(const __bf16* __restrict__ aG,
                                              const __bf16* __restrict__ bG,
                                              __bf16 (*As)[BM][LDK],
                                              __bf16 (*Bs)[BN][LDK],
                                              int blockM, int blockN,
                                              GemmAcc& acc) {
  const int tid  = threadIdx.x;
  const int wave = tid >> 5;
  const int wm   = wave & 3;          // 4 waves in M
  const int wn   = wave >> 2;         // 2 waves in N

  // each thread stages 2 b128 segments of A and 2 of B per K tile
  auto stage = [&](int buf, int k0) {
#pragma unroll
    for (int i = 0; i < 2; ++i) {
      const int s = tid + i * 256;           // 0..511
      const int row = s >> 2;                // 0..127
      const int ks  = (s & 3) * 8;           // 0,8,16,24
      async_b128_to_lds(aG + (size_t)(blockM + row) * C + k0 + ks, &As[buf][row][ks]);
      async_b128_to_lds(bG + (size_t)(blockN + row) * C + k0 + ks, &Bs[buf][row][ks]);
    }
  };

  stage(0, 0);
  wait_async0();
  __syncthreads();

  int buf = 0;
  for (int k0 = 0; k0 < C; k0 += BK) {
    if (k0 + BK < C) stage(buf ^ 1, k0 + BK);   // overlap with compute below
    v16bf a0 = load_frag_a(&As[buf][wm * 32 + 0][0],  LDK);
    v16bf a1 = load_frag_a(&As[buf][wm * 32 + 16][0], LDK);
#pragma unroll
    for (int jt = 0; jt < 4; ++jt) {
      v16bf bf = load_frag_b(&Bs[buf][wn * 64 + jt * 16][0], LDK);
      acc.a[0][jt] = WMMA_BF16(a0, bf, acc.a[0][jt]);
      acc.a[1][jt] = WMMA_BF16(a1, bf, acc.a[1][jt]);
    }
    wait_async0();        // next buffer landed (overlapped with ds+wmma above)
    __syncthreads();      // all waves done reading buf
    buf ^= 1;
  }
}

// ---- kernel 1: qkv = x @ w_qkv^T, fused ReLU + scatter --------------------
// q -> [B,H,N,D] (d contiguous), k,v -> [B,H,D,N] (n contiguous).
__global__ void __launch_bounds__(256) qkv_gemm(const __bf16* __restrict__ xb,
                                                const __bf16* __restrict__ wq,
                                                __bf16* __restrict__ qb,
                                                __bf16* __restrict__ kb,
                                                __bf16* __restrict__ vb) {
  __shared__ __bf16 As[2][BM][LDK];
  __shared__ __bf16 Bs[2][BN][LDK];

  const int lane = threadIdx.x & 31;
  const int wave = threadIdx.x >> 5;
  const int wm = wave & 3, wn = wave >> 2;
  const int blockM = blockIdx.x * BM;
  const int blockN = blockIdx.y * BN;

  GemmAcc acc = {};
  gemm_mainloop(xb, wq, As, Bs, blockM, blockN, acc);

  const int col  = lane & 15;
  const int mOff = (lane >> 4) * 8;
#pragma unroll
  for (int mf = 0; mf < 2; ++mf) {
    const int mBase = blockM + wm * 32 + mf * 16;
#pragma unroll
    for (int jt = 0; jt < 4; ++jt) {
      const int c3  = blockN + wn * 64 + jt * 16 + col;
      const int seg = c3 >> 10;        // 0=q 1=k 2=v
      const int cc  = c3 & 1023;
      const int h   = cc >> 6;
      const int d   = cc & 63;
      if (seg == 0) {                  // q: [b,h,n,d], strided scatter, ReLU
#pragma unroll
        for (int r = 0; r < 8; ++r) {
          const int t = mBase + mOff + r;
          const int b = t >> 12, n = t & (Ntok - 1);
          float f = acc.a[mf][jt][r];
          f = f > 0.f ? f : 0.f;
          qb[(((size_t)(b * H + h) * Ntok + n) << 6) + d] = (__bf16)f;
        }
      } else {                         // k,v: [b,h,d,n], 8 consecutive n -> b128
        __bf16* dst = (seg == 1) ? kb : vb;
        const bool relu = (seg == 1);
        const int t0 = mBase + mOff;
        const int b = t0 >> 12, n0 = t0 & (Ntok - 1);
        union { uint4 u; __bf16 hh[8]; } pk;
#pragma unroll
        for (int r = 0; r < 8; ++r) {
          float f = acc.a[mf][jt][r];
          if (relu) f = f > 0.f ? f : 0.f;
          pk.hh[r] = (__bf16)f;
        }
        *(uint4*)&dst[((size_t)(b * H + h) * Dh + d) * Ntok + n0] = pk.u;
      }
    }
  }
}

// ---- kernel 2: per-(b,h) linear attention core ----------------------------
__global__ void __launch_bounds__(256) attn_core(const __bf16* __restrict__ qb,
                                                 const __bf16* __restrict__ kb,
                                                 const __bf16* __restrict__ vb,
                                                 __bf16* __restrict__ attnb) {
  __shared__ float  zs[Ntok];        // 16 KB
  __shared__ float  ksum[Dh];
  __shared__ __bf16 ctx[Dh * Dh];    // 8 KB row-major [d][e]

  const int bh = blockIdx.x;
  const int b = bh >> 4, h = bh & 15;
  const __bf16* qh = qb + (size_t)bh * Ntok * Dh;   // [n][d]
  const __bf16* kh = kb + (size_t)bh * Dh * Ntok;   // [d][n]
  const __bf16* vh = vb + (size_t)bh * Dh * Ntok;   // [d][n]
  const int tid = threadIdx.x, wave = tid >> 5, lane = tid & 31;

  // phase 1: ksum[d] = sum_n k[d,n]
  if (tid < Dh) ksum[tid] = 0.f;
  __syncthreads();
  {
    const int d = tid >> 2, part = tid & 3;
    const __bf16* row = kh + (size_t)d * Ntok + part * 1024;
    float s = 0.f;
    for (int i = 0; i < 1024; i += 8) {
      union { uint4 u; __bf16 hh[8]; } pk;
      pk.u = *(const uint4*)(row + i);
#pragma unroll
      for (int j = 0; j < 8; ++j) s += (float)pk.hh[j];
    }
    atomicAdd(&ksum[d], s);          // ds_add_f32
  }
  __syncthreads();

  // phase 2: z[n] = 1/(ksum . q[n,:] + eps)
  for (int n = tid; n < Ntok; n += 256) {
    const __bf16* qrow = qh + (size_t)n * Dh;
    float dot = 0.f;
    for (int d0 = 0; d0 < Dh; d0 += 8) {
      union { uint4 u; __bf16 hh[8]; } pk;
      pk.u = *(const uint4*)(qrow + d0);
#pragma unroll
      for (int j = 0; j < 8; ++j) dot += ksum[d0 + j] * (float)pk.hh[j];
    }
    zs[n] = 1.f / (dot + 1.1920928955078125e-7f);
  }

  // phase 3: ctx[d][e] = sum_n v[d,n] k[e,n]; register double-buffered stream
  for (int t = wave; t < 16; t += 8) {
    const int dt = t >> 2, et = t & 3;
    v8f acc = {};
    const __bf16* aB = vh + (size_t)dt * 16 * Ntok;
    const __bf16* bB = kh + (size_t)et * 16 * Ntok;
    v16bf a  = load_frag_a(aB, Ntok);
    v16bf bf = load_frag_b(bB, Ntok);
    for (int n0 = BK; n0 < Ntok; n0 += BK) {
      v16bf an = load_frag_a(aB + n0, Ntok);     // issue next while WMMA runs
      v16bf bn = load_frag_b(bB + n0, Ntok);
      acc = WMMA_BF16(a, bf, acc);
      a = an; bf = bn;
    }
    acc = WMMA_BF16(a, bf, acc);
    const int col = lane & 15, mOff = (lane >> 4) * 8;
#pragma unroll
    for (int r = 0; r < 8; ++r)
      ctx[(dt * 16 + mOff + r) * Dh + et * 16 + col] = (__bf16)acc[r];
  }
  __syncthreads();

  // phase 4: out[d,n] = (ctx @ q^T)[d,n] * z[n] -> attnb[b*N+n][h*64+d]
  const int col = lane & 15, mOff = (lane >> 4) * 8;
  for (int dt = 0; dt < 4; ++dt) {
    v16bf a0 = load_frag_a(ctx + dt * 16 * Dh + 0,  Dh);   // ds_load
    v16bf a1 = load_frag_a(ctx + dt * 16 * Dh + 32, Dh);
    int nt = wave;
    v16bf b0 = load_frag_b(qh + (size_t)nt * 16 * Dh + 0,  Dh);
    v16bf b1 = load_frag_b(qh + (size_t)nt * 16 * Dh + 32, Dh);
    while (true) {
      const int ntn = nt + 8;
      v16bf c0, c1;
      if (ntn < Ntok / 16) {
        c0 = load_frag_b(qh + (size_t)ntn * 16 * Dh + 0,  Dh);
        c1 = load_frag_b(qh + (size_t)ntn * 16 * Dh + 32, Dh);
      }
      v8f acc = {};
      acc = WMMA_BF16(a0, b0, acc);
      acc = WMMA_BF16(a1, b1, acc);
      const int n = nt * 16 + col;
      const float z = zs[n];
      union { uint4 u; __bf16 hh[8]; } pk;
#pragma unroll
      for (int r = 0; r < 8; ++r) pk.hh[r] = (__bf16)(acc[r] * z);
      *(uint4*)&attnb[(size_t)(b * Ntok + n) * C + h * Dh + dt * 16 + mOff] = pk.u;
      if (ntn >= Ntok / 16) break;
      b0 = c0; b1 = c1; nt = ntn;
    }
  }
}

// ---- kernel 3: out = attn @ w_proj^T + b_proj (f32 output) ----------------
__global__ void __launch_bounds__(256) proj_gemm(const __bf16* __restrict__ ab,
                                                 const __bf16* __restrict__ wp,
                                                 const float* __restrict__ bias,
                                                 float* __restrict__ out) {
  __shared__ __bf16 As[2][BM][LDK];
  __shared__ __bf16 Bs[2][BN][LDK];

  const int lane = threadIdx.x & 31;
  const int wave = threadIdx.x >> 5;
  const int wm = wave & 3, wn = wave >> 2;
  const int blockM = blockIdx.x * BM;
  const int blockN = blockIdx.y * BN;

  GemmAcc acc = {};
  gemm_mainloop(ab, wp, As, Bs, blockM, blockN, acc);

  const int col = lane & 15, mOff = (lane >> 4) * 8;
#pragma unroll
  for (int mf = 0; mf < 2; ++mf) {
    const int mBase = blockM + wm * 32 + mf * 16;
#pragma unroll
    for (int jt = 0; jt < 4; ++jt) {
      const int j = blockN + wn * 64 + jt * 16 + col;
      const float bj = bias[j];
#pragma unroll
      for (int r = 0; r < 8; ++r)
        out[(size_t)(mBase + mOff + r) * C + j] = acc.a[mf][jt][r] + bj;
    }
  }
}

// ---------------------------------------------------------------------------
extern "C" void kernel_launch(void* const* d_in, const int* in_sizes, int n_in,
                              void* d_out, int out_size, void* d_ws, size_t ws_size,
                              hipStream_t stream) {
  (void)in_sizes; (void)n_in; (void)out_size; (void)ws_size;
  const float* x     = (const float*)d_in[0];
  const float* wqkv  = (const float*)d_in[1];
  const float* wproj = (const float*)d_in[2];
  const float* bproj = (const float*)d_in[3];
  float* out = (float*)d_out;

  char* ws = (char*)d_ws;
  size_t off = 0;
  auto alloc = [&](size_t bytes) -> char* {
    char* p = ws + off;
    off += (bytes + 255) & ~(size_t)255;
    return p;
  };
  __bf16* xb  = (__bf16*)alloc((size_t)MTOK * C * 2);
  __bf16* wqb = (__bf16*)alloc((size_t)C3 * C * 2);
  __bf16* wpb = (__bf16*)alloc((size_t)C * C * 2);
  __bf16* qb  = (__bf16*)alloc((size_t)MTOK * C * 2);   // [B,H,N,D]
  __bf16* kb  = (__bf16*)alloc((size_t)MTOK * C * 2);   // [B,H,D,N]
  __bf16* vb  = (__bf16*)alloc((size_t)MTOK * C * 2);   // [B,H,D,N]
  __bf16* atb = (__bf16*)alloc((size_t)MTOK * C * 2);   // [B*N, C]

  int n4;
  n4 = MTOK * C / 4;
  cvt_f32_bf16<<<(n4 + 255) / 256, 256, 0, stream>>>(x, xb, n4);
  n4 = C3 * C / 4;
  cvt_f32_bf16<<<(n4 + 255) / 256, 256, 0, stream>>>(wqkv, wqb, n4);
  n4 = C * C / 4;
  cvt_f32_bf16<<<(n4 + 255) / 256, 256, 0, stream>>>(wproj, wpb, n4);

  qkv_gemm<<<dim3(MTOK / BM, C3 / BN), 256, 0, stream>>>(xb, wqb, qb, kb, vb);
  attn_core<<<Bsz * H, 256, 0, stream>>>(qb, kb, vb, atb);
  proj_gemm<<<dim3(MTOK / BM, C / BN), 256, 0, stream>>>(atb, wpb, bproj, out);
}
